// FullyConnectedTensorProduct_47519518163610
// MI455X (gfx1250) — matrix operand order repaired and checked by reading, not compiled, not run on hardware
//
#include <hip/hip_runtime.h>

typedef __attribute__((ext_vector_type(16))) _Float16 v16h;
typedef __attribute__((ext_vector_type(8)))  float    v8f;

#define TP_MUL 256
#define INV_SQRT3 0.5773502691896258f

// ---------------------------------------------------------------------------
// Weight pack kernel: repack four 256x256 fp32 weight matrices (u-major:
// W[u*256 + w]) into f16 WMMA B-fragment order.
// B fragment order used here, contiguous k (V_WMMA_F32_16X16X32_F16, wave32):
//   lane L: half = L/16, Ncol = L%16 ; slot t (0..15): K = half*16 + t
// Packed layout: bp[((m*16 + ntile)*8 + kstep)*32 + lane][slot]  (f16)
// m: 0=w000, 1=w110, 2=w011, 3=w101
// ---------------------------------------------------------------------------
__global__ __launch_bounds__(256)
void tp_pack_weights(const float* __restrict__ w000, const float* __restrict__ w110,
                     const float* __restrict__ w011, const float* __restrict__ w101,
                     _Float16* __restrict__ bp) {
    int gid  = blockIdx.x * 256 + threadIdx.x;      // 0 .. 4*16*8*32*16-1
    int slot = gid & 15;
    int lane = (gid >> 4) & 31;
    int k    = (gid >> 9) & 7;
    int nt   = (gid >> 12) & 15;
    int m    = gid >> 16;
    int N    = (nt << 4) + (lane & 15);
    int u    = (k << 5) + ((lane >> 4) << 4) + slot;
    const float* W = (m == 0) ? w000 : (m == 1) ? w110 : (m == 2) ? w011 : w101;
    bp[gid] = (_Float16)W[u * 256 + N];
}

// ---------------------------------------------------------------------------
// Main fused kernel. One block = 16 rows. 8 waves, 32 tasks:
//   task 0..15  (S): out0 tile nt  -> A0@W000 (x s2) + A1@W110 (x 1/sqrt3)
//   task 16..31 (V): out1 tile nt  -> A0@W011 (x v2_i) + A{2+i}@W101 (x s2)
// A-fragment LDS layout (V_WMMA_F32_16X16X32_F16, wave32):
//   lane L: half = L/16, M = L%16
//   slot t: j=t/2, p=t%2 ; K = half*16 + 2*j + p
// ---------------------------------------------------------------------------
__global__ __launch_bounds__(256)
void tp_main(const float* __restrict__ x1, const float* __restrict__ x2,
             const _Float16* __restrict__ Bp, float* __restrict__ out, int nrows) {
    __shared__ _Float16 Apack[5 * 8 * 32 * 16];   // 40 KB: 5 A-matrices in frag order
    __shared__ float4   s2v2[16];                 // per-row {s2, v2x, v2y, v2z}

    const int tid = threadIdx.x;
    const int r0  = blockIdx.x * 16;

    // ---- stage x2 ----
    if (tid < 16) {
        int rr = r0 + tid;
        if (rr >= nrows) rr = nrows - 1;
        s2v2[tid] = ((const float4*)x2)[rr];
    }
    __syncthreads();

    // ---- build A fragments (5 derived matrices, 16 rows x 256 u) ----
    {
        const int u = tid;
        const int kstep = u >> 5;
        const int kl    = u & 31;
        const int half  = (kl >> 4) & 1;
        const int j     = (kl & 15) >> 1;
        const int p     = kl & 1;
        const int slot  = (j << 1) + p;
#pragma unroll 4
        for (int row = 0; row < 16; ++row) {
            int rr = r0 + row;
            if (rr >= nrows) rr = nrows - 1;
            const float* xr = x1 + (long)rr * 1024;
            float s1v = xr[u];
            const float* vp = xr + 256 + 3 * u;
            float vx = vp[0], vy = vp[1], vz = vp[2];
            float4 sv = s2v2[row];
            float dotv = vx * sv.y + vy * sv.z + vz * sv.w;

            int idx = (kstep * 32 + ((half << 4) + row)) * 16 + slot;
            Apack[idx               ] = (_Float16)s1v;   // A0 = s1
            Apack[idx + 1 * 8*32*16 ] = (_Float16)dotv;  // A1 = v1 . v2
            Apack[idx + 2 * 8*32*16 ] = (_Float16)vx;    // A2 = v1_x
            Apack[idx + 3 * 8*32*16 ] = (_Float16)vy;    // A3 = v1_y
            Apack[idx + 4 * 8*32*16 ] = (_Float16)vz;    // A4 = v1_z
        }
    }
    __syncthreads();

    // ---- GEMM phase ----
    const int wave = tid >> 5;
    const int lane = tid & 31;
    const int c    = lane & 15;     // output column within 16-tile
    const int half = lane >> 4;

    const v16h* Af = (const v16h*)Apack;   // 32B fragments
    const v16h* Bf = (const v16h*)Bp;

    const v8f zero = {0.f, 0.f, 0.f, 0.f, 0.f, 0.f, 0.f, 0.f};

    for (int t = 0; t < 4; ++t) {
        const int task = wave + (t << 3);
        const int nt   = task & 15;
        if (task < 16) {
            // ---- scalar-output task: out0 columns [nt*16, nt*16+16) ----
            v8f acc0 = zero, acc1 = zero;
#pragma unroll
            for (int k = 0; k < 8; ++k) {
                v16h a0 = Af[(0 * 8 + k) * 32 + lane];
                v16h a1 = Af[(1 * 8 + k) * 32 + lane];
                v16h b0 = Bf[((0 * 16 + nt) * 8 + k) * 32 + lane];
                v16h b1 = Bf[((1 * 16 + nt) * 8 + k) * 32 + lane];
                acc0 = __builtin_amdgcn_wmma_f32_16x16x32_f16(false, a0, false, b0,
                                                              (short)0, acc0, false, false);
                acc1 = __builtin_amdgcn_wmma_f32_16x16x32_f16(false, a1, false, b1,
                                                              (short)0, acc1, false, false);
            }
#pragma unroll
            for (int v = 0; v < 8; ++v) {
                int r = v + (half << 3);
                if (r0 + r < nrows) {
                    float4 sv = s2v2[r];
                    out[(long)(r0 + r) * 1024 + nt * 16 + c] =
                        sv.x * acc0[v] + INV_SQRT3 * acc1[v];
                }
            }
        } else {
            // ---- vector-output task: out1 w-columns [nt*16, nt*16+16), i=0..2 ----
            v8f acc011 = zero, accx = zero, accy = zero, accz = zero;
#pragma unroll
            for (int k = 0; k < 8; ++k) {
                v16h a0 = Af[(0 * 8 + k) * 32 + lane];
                v16h a2 = Af[(2 * 8 + k) * 32 + lane];
                v16h a3 = Af[(3 * 8 + k) * 32 + lane];
                v16h a4 = Af[(4 * 8 + k) * 32 + lane];
                v16h b2 = Bf[((2 * 16 + nt) * 8 + k) * 32 + lane];  // W011
                v16h b3 = Bf[((3 * 16 + nt) * 8 + k) * 32 + lane];  // W101 (shared x3)
                acc011 = __builtin_amdgcn_wmma_f32_16x16x32_f16(false, a0, false, b2,
                                                                (short)0, acc011, false, false);
                accx   = __builtin_amdgcn_wmma_f32_16x16x32_f16(false, a2, false, b3,
                                                                (short)0, accx, false, false);
                accy   = __builtin_amdgcn_wmma_f32_16x16x32_f16(false, a3, false, b3,
                                                                (short)0, accy, false, false);
                accz   = __builtin_amdgcn_wmma_f32_16x16x32_f16(false, a4, false, b3,
                                                                (short)0, accz, false, false);
            }
#pragma unroll
            for (int v = 0; v < 8; ++v) {
                int r = v + (half << 3);
                if (r0 + r < nrows) {
                    float4 sv = s2v2[r];
                    long ob = (long)(r0 + r) * 1024 + 256 + (long)(nt * 16 + c) * 3;
                    out[ob + 0] = INV_SQRT3 * (acc011[v] * sv.y + sv.x * accx[v]);
                    out[ob + 1] = INV_SQRT3 * (acc011[v] * sv.z + sv.x * accy[v]);
                    out[ob + 2] = INV_SQRT3 * (acc011[v] * sv.w + sv.x * accz[v]);
                }
            }
        }
    }
}

// ---------------------------------------------------------------------------
extern "C" void kernel_launch(void* const* d_in, const int* in_sizes, int n_in,
                              void* d_out, int out_size, void* d_ws, size_t ws_size,
                              hipStream_t stream) {
    // setup_inputs order: x1, x2, w000, w011, w101, w110
    const float* x1   = (const float*)d_in[0];
    const float* x2   = (const float*)d_in[1];
    const float* w000 = (const float*)d_in[2];
    const float* w011 = (const float*)d_in[3];
    const float* w101 = (const float*)d_in[4];
    const float* w110 = (const float*)d_in[5];

    const int nrows = in_sizes[0] / (4 * TP_MUL);   // 100000

    _Float16* Bp = (_Float16*)d_ws;                 // 4*16*8*32*16 f16 = 512 KB

    // pack weights into WMMA B-fragment order (deterministic, every call)
    tp_pack_weights<<<(4 * 16 * 8 * 32 * 16) / 256, 256, 0, stream>>>(
        w000, w110, w011, w101, Bp);

    // fused tensor-product kernel: one block per 16 rows
    tp_main<<<(nrows + 15) / 16, 256, 0, stream>>>(x1, x2, Bp, (float*)d_out, nrows);
}
